// MultiHeadSelfAttention_16733192585579
// MI455X (gfx1250) — compile-verified, matrix operand order checked by reading, not run on hardware
//
#include <hip/hip_runtime.h>
#include <hip/hip_bf16.h>

// ---------------------------------------------------------------------------
// Problem constants (from the reference)
// ---------------------------------------------------------------------------
#define BB 4
#define SS 1024
#define DD 1024
#define HH 16
#define HD 64
#define MM 128
#define NREL 257          // 2M+1
#define NRELP 272         // padded to 17*16
#define ROWS (BB*SS)      // 4096

typedef __attribute__((ext_vector_type(8)))  float  v8f;
typedef __attribute__((ext_vector_type(16))) __bf16 v16bf;

static __device__ inline __bf16 f2bf(float f) {
  unsigned u = __builtin_bit_cast(unsigned, f);
  u += 0x7fffu + ((u >> 16) & 1u);               // round-to-nearest-even
  unsigned short h = (unsigned short)(u >> 16);
  return __builtin_bit_cast(__bf16, h);
}

// Load a 16x32 bf16 WMMA A/B fragment: two contiguous 16-byte chunks.
static __device__ inline v16bf load_frag(const __bf16* p0, const __bf16* p1) {
  union { v16bf v; uint4 u[2]; } f;
  f.u[0] = *(const uint4*)p0;
  f.u[1] = *(const uint4*)p1;
  return f.v;
}

static __device__ inline v8f wmma_bf16(v16bf a, v16bf b, v8f c) {
  return __builtin_amdgcn_wmma_f32_16x16x32_bf16(false, a, false, b,
                                                 (short)0, c, false, false);
}

// CDNA5 async global->LDS DMA (ASYNCcnt-tracked), 16 bytes per lane.
static __device__ inline void async_ld_b128(unsigned lds_addr, unsigned voff,
                                            unsigned long long sbase) {
  asm volatile("global_load_async_to_lds_b128 %0, %1, %2"
               :: "v"(lds_addr), "v"(voff), "s"(sbase)
               : "memory");
}
static __device__ inline void wait_asynccnt0() {
  asm volatile("s_wait_asynccnt 0x0" ::: "memory");
}

// Row reductions across the 16-lane half-wave group (wave32).
static __device__ inline float rowmax16(float x) {
  for (int m = 1; m < 16; m <<= 1) x = fmaxf(x, __shfl_xor(x, m, 32));
  return x;
}
static __device__ inline float rowsum16(float x) {
  for (int m = 1; m < 16; m <<= 1) x += __shfl_xor(x, m, 32);
  return x;
}

// ---------------------------------------------------------------------------
// Elementwise f32 -> bf16
// ---------------------------------------------------------------------------
__global__ void cvt_bf16_kernel(const float* __restrict__ src,
                                __bf16* __restrict__ dst, int n) {
  int i = blockIdx.x * 256 + threadIdx.x;
  if (i < n) dst[i] = f2bf(src[i]);
}

// W [K=1024, N=1024] f32 (row-major)  ->  Wt [N, K] bf16 (row-major)
__global__ void transpose_bf16_kernel(const float* __restrict__ W,
                                      __bf16* __restrict__ Wt) {
  int n = blockIdx.x * 16 + threadIdx.x;
  int k = blockIdx.y * 16 + threadIdx.y;
  Wt[(size_t)n * DD + k] = f2bf(W[(size_t)k * DD + n]);
}

// Sinusoidal relative-position table, row-major [NRELP][HD] bf16, rows >=257 = 0
__global__ void relemb_kernel(__bf16* __restrict__ Rel) {
  int d = threadIdx.x;          // 0..63
  int r = blockIdx.x;           // 0..271
  float val = 0.0f;
  if (r < NREL) {
    float rel = (float)(r - MM);
    int i = d >> 1;
    float inv = __expf(-((float)(2 * i) / (float)HD) * logf(10000.0f));
    float ang = rel * inv;
    val = (d & 1) ? __cosf(ang) : __sinf(ang);
  }
  Rel[(size_t)r * HD + d] = f2bf(val);
}

// ---------------------------------------------------------------------------
// WMMA GEMM: C[4096,1024] = A(bf16,[4096,1024]) * Bt(bf16,[1024,1024]^T) + bias
// 128 threads = 4 waves (2x2), wave tile 64x64, WG tile 128x128.
// A/B 128x32 tiles staged in LDS via global_load_async_to_lds_b128, double
// buffered; fragments read from LDS as contiguous B128 chunks.
// MODE 0: write (c+bq+u) and (c+bq+v) to Qu/Qv in [B,H,S,HD] bf16
// MODE 1: write (c+bk)           to K     in [B,H,S,HD] bf16
// MODE 2: write (c+bv)           to Vt    in [B,H,HD,S] bf16 (packed B128)
// MODE 3: write (c+bo)           to f32 out row-major
// ---------------------------------------------------------------------------
template <int MODE>
__global__ __launch_bounds__(128)
void gemm_wmma_kernel(const __bf16* __restrict__ A, const __bf16* __restrict__ Bt,
                      const float* __restrict__ bias, const float* __restrict__ uvec,
                      const float* __restrict__ vvec, __bf16* __restrict__ out0,
                      __bf16* __restrict__ out1, float* __restrict__ outf) {
  __shared__ __align__(16) __bf16 bufA[2][128 * 32];
  __shared__ __align__(16) __bf16 bufB[2][128 * 32];

  const int lane = threadIdx.x & 31;
  const int w    = threadIdx.x >> 5;
  const int hf   = lane >> 4;          // half-wave
  const int li   = lane & 15;
  const int mBase = blockIdx.y * 128;
  const int nBase = blockIdx.x * 128;
  const int mW = (w >> 1) * 64;        // wave tile within WG tile
  const int nW = (w & 1) * 64;

  // Issue one 128x32 A tile + 128x32 B tile as async DMA into LDS buffer `buf`.
  auto issue_tile = [&](int kk, int buf) {
    const unsigned t = threadIdx.x;            // one row per thread
    unsigned long long baseA =
        (unsigned long long)(A + (size_t)mBase * DD + kk);
    unsigned long long baseB =
        (unsigned long long)(Bt + (size_t)nBase * DD + kk);
    for (int c = 0; c < 4; ++c) {              // 4 x 16B = one 64B row
      unsigned voff = (t * DD + c * 8) * 2;    // byte offset in global
      async_ld_b128((unsigned)(size_t)&bufA[buf][t * 32 + c * 8], voff, baseA);
      async_ld_b128((unsigned)(size_t)&bufB[buf][t * 32 + c * 8], voff, baseB);
    }
  };

  v8f acc[4][4];
  for (int i = 0; i < 4; ++i)
    for (int j = 0; j < 4; ++j) acc[i][j] = {};

  issue_tile(0, 0);
  wait_asynccnt0();
  __syncthreads();

  for (int kk = 0; kk < DD; kk += 32) {
    const int cur = (kk >> 5) & 1;
    if (kk + 32 < DD) issue_tile(kk + 32, cur ^ 1);

    v16bf af[4], bfr[4];
    for (int i = 0; i < 4; ++i) {
      const __bf16* p = &bufA[cur][(mW + i * 16 + li) * 32];
      af[i] = load_frag(p + hf * 8, p + 16 + hf * 8);
    }
    for (int j = 0; j < 4; ++j) {
      const __bf16* p = &bufB[cur][(nW + j * 16 + li) * 32];
      bfr[j] = load_frag(p + hf * 8, p + 16 + hf * 8);
    }
    for (int i = 0; i < 4; ++i)
      for (int j = 0; j < 4; ++j)
        acc[i][j] = wmma_bf16(af[i], bfr[j], acc[i][j]);

    wait_asynccnt0();   // next buffer's DMA has landed (overlapped with math)
    __syncthreads();    // visible to all waves; reads of `cur` finished
  }

  // Epilogue. C layout: lane col = n_tile + li ; VGPR v row = m_tile + v + 8*hf.
  const int m0 = mBase + mW;
  const int n0 = nBase + nW;
  for (int i = 0; i < 4; ++i) {
    for (int j = 0; j < 4; ++j) {
      const int col = n0 + j * 16 + li;
      const float bvl = bias[col];
      const int h  = col >> 6, hd = col & 63;
      if (MODE == 2) {
        const int rowbase = m0 + i * 16 + 8 * hf;   // 8 consecutive rows
        const int b = rowbase >> 10, s = rowbase & (SS - 1);
        union { uint4 u; __bf16 hx[8]; } pk;
        for (int vv = 0; vv < 8; ++vv) pk.hx[vv] = f2bf(acc[i][j][vv] + bvl);
        size_t idx = ((size_t)((b * HH + h) * HD + hd)) * SS + s;
        *(uint4*)&out0[idx] = pk.u;
      } else {
        for (int vv = 0; vv < 8; ++vv) {
          const int row = m0 + i * 16 + vv + 8 * hf;
          const float val = acc[i][j][vv] + bvl;
          if (MODE == 3) {
            outf[(size_t)row * DD + col] = val;
          } else {
            const int b = row >> 10, s = row & (SS - 1);
            size_t idx = ((size_t)((b * HH + h) * SS + s)) * HD + hd;
            if (MODE == 0) {
              out0[idx] = f2bf(val + uvec[col]);
              out1[idx] = f2bf(val + vvec[col]);
            } else {
              out0[idx] = f2bf(val);
            }
          }
        }
      }
    }
  }
}

// ---------------------------------------------------------------------------
// Flash attention with exact relative-position scores.
// Grid (B*H, S/64); 128 threads = 4 waves; wave owns 16 query rows.
// ---------------------------------------------------------------------------
__global__ __launch_bounds__(128)
void attn_kernel(const __bf16* __restrict__ Qu, const __bf16* __restrict__ Qv,
                 const __bf16* __restrict__ Kb, const __bf16* __restrict__ Vt,
                 const __bf16* __restrict__ Rel, __bf16* __restrict__ Out) {
  __shared__ float ldsRel[64 * NRELP];                  // pre-scaled rel scores
  __shared__ __align__(16) __bf16 pStage[4][16 * 32];   // per-wave P re-layout

  const int bh   = blockIdx.x;          // b*16 + h
  const int qblk = blockIdx.y;          // 0..15
  const int lane = threadIdx.x & 31;
  const int w    = threadIdx.x >> 5;
  const int hf   = lane >> 4;
  const int li   = lane & 15;
  const int iBase = qblk * 64 + w * 16; // wave's first query row (s index)
  const float scale = 0.125f;           // 1/sqrt(64)

  // Load persistent q+u / q+v A-fragments for this wave's 16 rows.
  const __bf16* qr  = Qu + ((size_t)bh * SS + iBase + li) * HD;
  const __bf16* qvr = Qv + ((size_t)bh * SS + iBase + li) * HD;
  v16bf quA0 = load_frag(qr  + hf * 8, qr  + 16 + hf * 8);
  v16bf quA1 = load_frag(qr  + 32 + hf * 8, qr + 48 + hf * 8);
  v16bf qvA0 = load_frag(qvr + hf * 8, qvr + 16 + hf * 8);
  v16bf qvA1 = load_frag(qvr + 32 + hf * 8, qvr + 48 + hf * 8);

  // Phase 1: rel table for this wave's rows: (q+v) @ rel_emb^T * scale -> LDS
  for (int t = 0; t < NRELP / 16; ++t) {
    const __bf16* rb = Rel + (size_t)(t * 16 + li) * HD;
    v16bf b0 = load_frag(rb + hf * 8, rb + 16 + hf * 8);
    v16bf b1 = load_frag(rb + 32 + hf * 8, rb + 48 + hf * 8);
    v8f c = {};
    c = wmma_bf16(qvA0, b0, c);
    c = wmma_bf16(qvA1, b1, c);
    for (int vv = 0; vv < 8; ++vv)
      ldsRel[(w * 16 + vv + 8 * hf) * NRELP + t * 16 + li] = c[vv] * scale;
  }
  __syncthreads();

  float mrow[8], lrow[8];
  for (int vv = 0; vv < 8; ++vv) { mrow[vv] = -1e30f; lrow[vv] = 0.0f; }
  v8f o[4];
  for (int nt = 0; nt < 4; ++nt) o[nt] = {};

  const __bf16* Kbase = Kb + (size_t)bh * SS * HD;
  const __bf16* Vbase = Vt + (size_t)bh * HD * SS;
  __bf16* ps = pStage[w];

  for (int j0 = 0; j0 < SS; j0 += 32) {
    // content scores, two 16-col tiles
    v8f sc[2];
    for (int t = 0; t < 2; ++t) {
      const __bf16* kr = Kbase + (size_t)(j0 + t * 16 + li) * HD;
      v16bf kb0 = load_frag(kr + hf * 8, kr + 16 + hf * 8);
      v16bf kb1 = load_frag(kr + 32 + hf * 8, kr + 48 + hf * 8);
      v8f c = {};
      c = wmma_bf16(quA0, kb0, c);
      c = wmma_bf16(quA1, kb1, c);
      sc[t] = c;
    }

    // scale + exact rel gather + running max
    float mstep[8];
    for (int vv = 0; vv < 8; ++vv) {
      const int iloc = w * 16 + vv + 8 * hf;
      const int ig   = qblk * 64 + iloc;
      int r0 = (j0 + li) - ig;       r0 = r0 < -MM ? -MM : (r0 > MM ? MM : r0);
      int r1 = (j0 + 16 + li) - ig;  r1 = r1 < -MM ? -MM : (r1 > MM ? MM : r1);
      float s0 = sc[0][vv] * scale + ldsRel[iloc * NRELP + r0 + MM];
      float s1 = sc[1][vv] * scale + ldsRel[iloc * NRELP + r1 + MM];
      sc[0][vv] = s0; sc[1][vv] = s1;
      mstep[vv] = rowmax16(fmaxf(s0, s1));
    }
    float fac[8];
    for (int vv = 0; vv < 8; ++vv) {
      float mn = fmaxf(mrow[vv], mstep[vv]);
      fac[vv] = __expf(mrow[vv] - mn);
      mrow[vv] = mn;
    }

    // P = exp(s - m), stage to LDS (C layout -> A layout), row sums
    for (int vv = 0; vv < 8; ++vv) {
      float p0 = __expf(sc[0][vv] - mrow[vv]);
      float p1 = __expf(sc[1][vv] - mrow[vv]);
      int roff = (vv + 8 * hf) * 32;
      ps[roff + li]      = f2bf(p0);
      ps[roff + 16 + li] = f2bf(p1);
      lrow[vv] = lrow[vv] * fac[vv] + rowsum16(p0 + p1);
    }

    // read P back as a 16x32 A fragment (DS pipe is in-order per wave)
    const __bf16* pr = ps + li * 32;
    v16bf pA = load_frag(pr + hf * 8, pr + 16 + hf * 8);

    // rescale O and accumulate P @ V (Vt gives contiguous B-fragments)
    for (int nt = 0; nt < 4; ++nt) {
      for (int vv = 0; vv < 8; ++vv) o[nt][vv] *= fac[vv];
      const __bf16* vr = Vbase + (size_t)(nt * 16 + li) * SS + j0;
      v16bf vb = load_frag(vr + hf * 8, vr + 16 + hf * 8);
      o[nt] = wmma_bf16(pA, vb, o[nt]);
    }
  }

  // normalize and write [B,S,D] bf16 for the output projection
  const int b = bh >> 4, h = bh & 15;
  for (int nt = 0; nt < 4; ++nt) {
    const int hd = nt * 16 + li;
    for (int vv = 0; vv < 8; ++vv) {
      const int s = qblk * 64 + w * 16 + vv + 8 * hf;
      float val = o[nt][vv] / lrow[vv];
      Out[((size_t)(b * SS + s)) * DD + h * HD + hd] = f2bf(val);
    }
  }
}

// ---------------------------------------------------------------------------
// Host-side launcher
// ---------------------------------------------------------------------------
extern "C" void kernel_launch(void* const* d_in, const int* in_sizes, int n_in,
                              void* d_out, int out_size, void* d_ws, size_t ws_size,
                              hipStream_t stream) {
  (void)in_sizes; (void)n_in; (void)out_size; (void)ws_size;
  const float* x  = (const float*)d_in[0];
  const float* Wq = (const float*)d_in[1];
  const float* bq = (const float*)d_in[2];
  const float* Wk = (const float*)d_in[3];
  const float* bk = (const float*)d_in[4];
  const float* Wv = (const float*)d_in[5];
  const float* bv = (const float*)d_in[6];
  const float* Wo = (const float*)d_in[7];
  const float* bo = (const float*)d_in[8];
  const float* uP = (const float*)d_in[9];   // [H,HD] flat == indexed by col
  const float* vP = (const float*)d_in[10];
  float* out = (float*)d_out;

  char* ws = (char*)d_ws;
  auto alloc = [&](size_t bytes) {
    char* p = ws;
    ws += (bytes + 255) & ~(size_t)255;
    return p;
  };
  __bf16* xbf   = (__bf16*)alloc((size_t)ROWS * DD * 2);
  __bf16* WqT   = (__bf16*)alloc((size_t)DD * DD * 2);
  __bf16* WkT   = (__bf16*)alloc((size_t)DD * DD * 2);
  __bf16* WvT   = (__bf16*)alloc((size_t)DD * DD * 2);
  __bf16* WoT   = (__bf16*)alloc((size_t)DD * DD * 2);
  __bf16* RelE  = (__bf16*)alloc((size_t)NRELP * HD * 2);
  __bf16* Qu    = (__bf16*)alloc((size_t)ROWS * DD * 2);
  __bf16* Qv    = (__bf16*)alloc((size_t)ROWS * DD * 2);
  __bf16* Kbuf  = (__bf16*)alloc((size_t)ROWS * DD * 2);
  __bf16* Vt    = (__bf16*)alloc((size_t)ROWS * DD * 2);
  __bf16* AttnO = (__bf16*)alloc((size_t)ROWS * DD * 2);

  const int nElem = ROWS * DD;  // 4M
  cvt_bf16_kernel<<<(nElem + 255) / 256, 256, 0, stream>>>(x, xbf, nElem);

  dim3 tg(64, 64), tb(16, 16);
  transpose_bf16_kernel<<<tg, tb, 0, stream>>>(Wq, WqT);
  transpose_bf16_kernel<<<tg, tb, 0, stream>>>(Wk, WkT);
  transpose_bf16_kernel<<<tg, tb, 0, stream>>>(Wv, WvT);
  transpose_bf16_kernel<<<tg, tb, 0, stream>>>(Wo, WoT);

  relemb_kernel<<<NRELP, HD, 0, stream>>>(RelE);

  dim3 gg(DD / 128, ROWS / 128);  // (8, 32)
  gemm_wmma_kernel<0><<<gg, 128, 0, stream>>>(xbf, WqT, bq, uP, vP, Qu, Qv, nullptr);
  gemm_wmma_kernel<1><<<gg, 128, 0, stream>>>(xbf, WkT, bk, nullptr, nullptr, Kbuf, nullptr, nullptr);
  gemm_wmma_kernel<2><<<gg, 128, 0, stream>>>(xbf, WvT, bv, nullptr, nullptr, Vt, nullptr, nullptr);

  attn_kernel<<<dim3(BB * HH, SS / 64), 128, 0, stream>>>(Qu, Qv, Kbuf, Vt, RelE, AttnO);

  gemm_wmma_kernel<3><<<gg, 128, 0, stream>>>(AttnO, WoT, bo, nullptr, nullptr, nullptr, nullptr, out);
}
